// QuantumFlowMatching_20899310862992
// MI455X (gfx1250) — matrix operand order repaired and checked by reading, not compiled, not run on hardware
//
#include <hip/hip_runtime.h>
#include <hip/hip_fp16.h>

// ---------------------------------------------------------------------------
// 14-qubit batched statevector sim for MI455X (gfx1250, wave32).
// Strategy: state (2^14 c64 = 128KB) lives entirely in LDS (320KB/WGP),
// one workgroup per batch element; WMMA f32 16x16x4 for the encoding GEMM,
// WMMA f32 16x16x32 f16 for the head GEMM; async global->LDS staging for
// the per-gate SU(4) matrices and the 16x16 observables (ASYNCcnt path).
// ---------------------------------------------------------------------------

#define NQ      14
#define BATCH   128
#define NGATES  26
#define NOBS    11
#define ENC     390
#define INDIM   784
#define OUTDIM  10
#define NOFF    120

typedef float    v2f  __attribute__((ext_vector_type(2)));
typedef float    v8f  __attribute__((ext_vector_type(8)));
typedef _Float16 v16h __attribute__((ext_vector_type(16)));

// SU4 brick order: even pairs then odd pairs, 2 blocks.
__constant__ int PAIRQ[NGATES] = {0,2,4,6,8,10,12, 1,3,5,7,9,11,
                                  0,2,4,6,8,10,12, 1,3,5,7,9,11};

// ---------------------------------------------------------------------------
// Async global->LDS copy (CDNA5 ASYNCcnt path). Dest VGPR holds the LDS byte
// offset (generic -> addrspace(3) cast), source is a per-lane 64-bit address.
// ---------------------------------------------------------------------------
typedef __attribute__((address_space(3))) char lds_char_t;

__device__ __forceinline__ void async_copy_b64(void* lds_dst, const void* gsrc) {
  const unsigned dst = (unsigned)(unsigned long long)(lds_char_t*)lds_dst;
  asm volatile("global_load_async_to_lds_b64 %0, %1, off"
               :: "v"(dst), "v"(gsrc) : "memory");
}
__device__ __forceinline__ void async_wait0() {
  asm volatile("s_wait_asynccnt 0x0" ::: "memory");
}

// ---------------------------------------------------------------------------
// Kernel 1: encoded = x @ W_enc^T + b_enc  (128 x 784 x 390), f32 WMMA.
// One wave per 16x16 output tile; K stepped by 4 via V_WMMA_F32_16X16X4_F32.
// A layout (16x4 f32): lane l holds M=l%16; comp0->K=k0+(l>=16?2:0), comp1->+1.
// B layout (4x16 f32): lane l holds N=l%16; same K split (symmetric to A).
// C/D layout: VGPR v -> M = v + (l>=16 ? 8:0), N = l%16.
// ---------------------------------------------------------------------------
__global__ __launch_bounds__(32) void enc_gemm(const float* __restrict__ X,
                                               const float* __restrict__ W,
                                               const float* __restrict__ bias,
                                               float* __restrict__ enc) {
  const int lane = threadIdx.x;
  const int m0 = blockIdx.x * 16;
  const int n0 = blockIdx.y * 16;
  const int row  = m0 + (lane & 15);
  const int col  = n0 + (lane & 15);
  const int colc = (col < ENC) ? col : (ENC - 1);
  const int khalf = (lane >> 4) << 1;              // 0 or 2
  const float* xrow = X + row * INDIM + khalf;
  const float* wrow = W + colc * INDIM + khalf;
  v8f acc = {};
  for (int k0 = 0; k0 < INDIM; k0 += 4) {
    if ((k0 & 63) == 0 && k0 + 64 < INDIM) {       // global_prefetch_b8
      __builtin_prefetch(xrow + k0 + 64, 0, 0);
      __builtin_prefetch(wrow + k0 + 64, 0, 0);
    }
    const float2 af = *(const float2*)(xrow + k0);
    const float2 bf = *(const float2*)(wrow + k0);
    v2f a; a.x = af.x; a.y = af.y;
    v2f b; b.x = bf.x; b.y = bf.y;
    acc = __builtin_amdgcn_wmma_f32_16x16x4_f32(false, a, false, b,
                                                (short)0, acc, false, false);
  }
  if (col < ENC) {
    const float bv = bias[col];
    const int mrow = m0 + ((lane >> 4) << 3);
#pragma unroll
    for (int v = 0; v < 8; ++v)
      enc[(mrow + v) * ENC + col] = acc[v] + bv;
  }
}

// ---------------------------------------------------------------------------
// Kernel 2: U_g = exp(i * sum_m theta_m P_m) for all (gate, batch).
// Fully unrolled Pauli-word accumulation + scaling-and-squaring Taylor.
// ---------------------------------------------------------------------------
__device__ __forceinline__ void pauli1(int p, int r, int& c, float& pr, float& pi) {
  switch (p) {
    case 0:  c = r;     pr = 1.f;              pi = 0.f;              break; // I
    case 1:  c = r ^ 1; pr = 1.f;              pi = 0.f;              break; // X
    case 2:  c = r ^ 1; pr = 0.f;              pi = r ? 1.f : -1.f;   break; // Y
    default: c = r;     pr = r ? -1.f : 1.f;   pi = 0.f;              break; // Z
  }
}

__device__ __forceinline__ void cmm4(const float* ar, const float* ai,
                                     const float* br, const float* bi,
                                     float* cr, float* ci) {
#pragma unroll
  for (int r = 0; r < 4; ++r)
#pragma unroll
    for (int c = 0; c < 4; ++c) {
      float sr = 0.f, si = 0.f;
#pragma unroll
      for (int k = 0; k < 4; ++k) {
        const float xr = ar[r*4+k], xi = ai[r*4+k];
        const float yr = br[k*4+c], yi = bi[k*4+c];
        sr += xr*yr - xi*yi;
        si += xr*yi + xi*yr;
      }
      cr[r*4+c] = sr; ci[r*4+c] = si;
    }
}

__global__ __launch_bounds__(128) void su4_kernel(const float* __restrict__ enc,
                                                  float2* __restrict__ Ug) {
  const int b = threadIdx.x;   // batch
  const int g = blockIdx.x;    // gate
  float th[15];
#pragma unroll
  for (int m = 0; m < 15; ++m) th[m] = enc[b * ENC + g * 15 + m];

  float mr[16] = {}, mi[16] = {};      // M = i*H
  float nrm = 0.f;
#pragma unroll
  for (int m = 0; m < 15; ++m) {
    nrm += fabsf(th[m]);
    const int word = m + 1, pa = word >> 2, pb = word & 3;
#pragma unroll
    for (int r = 0; r < 4; ++r) {
      int c1, c0; float a_r, a_i, b_r, b_i;
      pauli1(pa, r >> 1, c1, a_r, a_i);
      pauli1(pb, r & 1,  c0, b_r, b_i);
      const int   c  = c1 * 2 + c0;
      const float pr = a_r * b_r - a_i * b_i;
      const float pi = a_r * b_i + a_i * b_r;
      // theta * i * (pr + i*pi) = theta * (-pi + i*pr)
      mr[r*4+c] += th[m] * (-pi);
      mi[r*4+c] += th[m] * pr;
    }
  }
  // ||H||_2 <= sum |theta_m|  (unit-spectral-norm Pauli words)
  int s = 0; float sc = 1.f;
  while (nrm > 0.5f && s < 24) { nrm *= 0.5f; sc *= 0.5f; ++s; }
#pragma unroll
  for (int i = 0; i < 16; ++i) { mr[i] *= sc; mi[i] *= sc; }

  float ur[16] = {}, ui[16] = {}, tr[16] = {}, ti[16] = {};
  float t2r[16], t2i[16];
#pragma unroll
  for (int i = 0; i < 4; ++i) { ur[i*5] = 1.f; tr[i*5] = 1.f; }
  for (int k = 1; k <= 12; ++k) {                // Taylor to ~1e-9 at ||M||<0.5
    cmm4(tr, ti, mr, mi, t2r, t2i);
    const float inv = 1.f / (float)k;
#pragma unroll
    for (int i = 0; i < 16; ++i) {
      tr[i] = t2r[i] * inv; ti[i] = t2i[i] * inv;
      ur[i] += tr[i];       ui[i] += ti[i];
    }
  }
  for (int it = 0; it < s; ++it) {               // repeated squaring
    cmm4(ur, ui, ur, ui, t2r, t2i);
#pragma unroll
    for (int i = 0; i < 16; ++i) { ur[i] = t2r[i]; ui[i] = t2i[i]; }
  }
  float2* o = Ug + (g * BATCH + b) * 16;
#pragma unroll
  for (int i = 0; i < 16; ++i) o[i] = make_float2(ur[i], ui[i]);
}

// ---------------------------------------------------------------------------
// Kernel 3: build the 11 Hermitian 16x16 observables (h + h^dagger).
// ---------------------------------------------------------------------------
__global__ __launch_bounds__(256) void herm_kernel(const float* __restrict__ A,
                                                   const float* __restrict__ Bp,
                                                   const float* __restrict__ D,
                                                   float2* __restrict__ Hs) {
  const int w = blockIdx.x, t = threadIdx.x;
  const int i = t >> 4, j = t & 15;
  float re, im;
  if (i > j)      { const int lin = i*(i-1)/2 + j; re = A[w*NOFF+lin]; im =  Bp[w*NOFF+lin]; }
  else if (i < j) { const int lin = j*(j-1)/2 + i; re = A[w*NOFF+lin]; im = -Bp[w*NOFF+lin]; }
  else            { re = (i < 15) ? 2.f * D[w*16 + i + 1] : 0.f; im = 0.f; }
  Hs[w*256 + t] = make_float2(re, im);
}

// ---------------------------------------------------------------------------
// Kernel 4: LDS-resident statevector circuit. One WG per batch element.
// Dynamic LDS: 16384 float2 state + 256 float2 scratch (U / H) + 256 float.
// Staging tiles go global->LDS via the async engine (ASYNCcnt), no VGPR hop.
// ---------------------------------------------------------------------------
#define SIM_THREADS 256
#define SIM_SMEM ((16384 + 256) * sizeof(float2) + 256 * sizeof(float))

__global__ __launch_bounds__(SIM_THREADS) void sim_kernel(
    const float2* __restrict__ Ug, const float* __restrict__ vp,
    const float2* __restrict__ Hs, float* __restrict__ qout) {
  extern __shared__ char smem_raw[];
  float2* st  = (float2*)smem_raw;                 // 16384 amplitudes
  float2* scr = st + 16384;                        // 256: U (16) or H (256)
  float*  red = (float*)(scr + 256);               // 256: reduction

  const int b = blockIdx.x, tid = threadIdx.x;
  for (int i = tid; i < (1 << NQ); i += SIM_THREADS) st[i] = make_float2(0.f, 0.f);
  __syncthreads();
  if (tid == 0) st[0] = make_float2(1.f, 0.f);
  __syncthreads();

  // ---- SU(4) brick-layer encoding ----
  for (int g = 0; g < NGATES; ++g) {
    const int q = PAIRQ[g];
    if (tid < 16)                                  // async stage U (16 x c64)
      async_copy_b64(&scr[tid], &Ug[(g * BATCH + b) * 16 + tid]);
    async_wait0();
    __syncthreads();
    const int pL = 12 - q;                         // bit pos of wire q+1
    const unsigned mk = (1u << pL) - 1u;
    for (int grp = tid; grp < 4096; grp += SIM_THREADS) {
      const unsigned base = ((unsigned)(grp >> pL) << (pL + 2)) | ((unsigned)grp & mk);
      const float2 a0 = st[base];
      const float2 a1 = st[base + (1u << pL)];
      const float2 a2 = st[base + (2u << pL)];
      const float2 a3 = st[base + (3u << pL)];
#pragma unroll
      for (int k = 0; k < 4; ++k) {
        const float2 u0 = scr[k*4+0], u1 = scr[k*4+1];
        const float2 u2 = scr[k*4+2], u3 = scr[k*4+3];
        const float orr = u0.x*a0.x - u0.y*a0.y + u1.x*a1.x - u1.y*a1.y
                        + u2.x*a2.x - u2.y*a2.y + u3.x*a3.x - u3.y*a3.y;
        const float oii = u0.x*a0.y + u0.y*a0.x + u1.x*a1.y + u1.y*a1.x
                        + u2.x*a2.y + u2.y*a2.x + u3.x*a3.y + u3.y*a3.x;
        st[base + ((unsigned)k << pL)] = make_float2(orr, oii);
      }
    }
    __syncthreads();
  }

  // ---- variational RY + CNOT brick layers ----
  for (int layer = 0; layer < 3; ++layer) {
    for (int q = 0; q < NQ; ++q) {
      const float th = vp[layer * NQ + q] * 0.5f;
      const float c = cosf(th), s = sinf(th);
      const int pq = 13 - q;
      const unsigned mq = (1u << pq) - 1u;
      for (int p = tid; p < 8192; p += SIM_THREADS) {
        const unsigned base = ((unsigned)(p >> pq) << (pq + 1)) | ((unsigned)p & mq);
        const float2 a0 = st[base], a1 = st[base + (1u << pq)];
        st[base]             = make_float2(c*a0.x - s*a1.x, c*a0.y - s*a1.y);
        st[base + (1u<<pq)]  = make_float2(s*a0.x + c*a1.x, s*a0.y + c*a1.y);
      }
      __syncthreads();
    }
#pragma unroll
    for (int half = 0; half < 2; ++half) {
      for (int q = half; q < NQ - 1; q += 2) {     // CNOT(q, q+1): swap |10>,|11>
        const int pL = 12 - q;
        const unsigned mk = (1u << pL) - 1u;
        for (int grp = tid; grp < 4096; grp += SIM_THREADS) {
          const unsigned base = ((unsigned)(grp >> pL) << (pL + 2)) | ((unsigned)grp & mk);
          const unsigned i2 = base + (2u << pL), i3 = base + (3u << pL);
          const float2 t = st[i2]; st[i2] = st[i3]; st[i3] = t;
        }
        __syncthreads();
      }
    }
  }

  // ---- expectation values: <psi| H_w |psi> on wires [w..w+3] ----
  for (int w = 0; w < NOBS; ++w) {
    async_copy_b64(&scr[tid], &Hs[w * 256 + tid]);  // async stage H (256 x c64)
    async_wait0();
    __syncthreads();
    const int sh = 10 - w;
    const unsigned mrm = (1u << sh) - 1u;
    float local = 0.f;
    for (int r = tid; r < 1024; r += SIM_THREADS) {
      const unsigned base = ((unsigned)(r >> sh) << (sh + 4)) | ((unsigned)r & mrm);
      float2 v[16];
#pragma unroll
      for (int k = 0; k < 16; ++k) v[k] = st[base + ((unsigned)k << sh)];
#pragma unroll
      for (int k = 0; k < 16; ++k) {
        float tr_ = 0.f, ti_ = 0.f;
#pragma unroll
        for (int l = 0; l < 16; ++l) {
          const float2 h = scr[k*16+l];
          tr_ += h.x * v[l].x - h.y * v[l].y;
          ti_ += h.x * v[l].y + h.y * v[l].x;
        }
        local += v[k].x * tr_ + v[k].y * ti_;      // Re(conj(v_k) * t_k)
      }
    }
    red[tid] = local;
    __syncthreads();
    for (int off = 128; off > 0; off >>= 1) {
      if (tid < off) red[tid] += red[tid + off];
      __syncthreads();
    }
    if (tid == 0) qout[b * NOBS + w] = red[0];
    __syncthreads();
  }
}

// ---------------------------------------------------------------------------
// Kernel 5: head GEMM out = qout @ W_head^T + b_head via f16 WMMA (K padded
// 11->32, N padded 10->16). Tolerance-insensitive tiny linear layer.
// ---------------------------------------------------------------------------
__global__ __launch_bounds__(32) void head_kernel(const float* __restrict__ qo,
                                                  const float* __restrict__ Wh,
                                                  const float* __restrict__ bh,
                                                  float* __restrict__ out) {
  const int lane = threadIdx.x;
  const int m0 = blockIdx.x * 16;
  const int nn = lane & 15;
  const int hi = lane >> 4;
  const int m = m0 + nn;
  v16h a, bm;
#pragma unroll
  for (int j = 0; j < 8; ++j) {   // A 16x32 f16 layout: paired K per VGPR
    const int kb = ((j < 4) ? (2 * j) : (2 * (j - 4) + 16)) + (hi ? 8 : 0);
    a[2*j]   = (kb     < NOBS) ? (_Float16)qo[m * NOBS + kb]     : (_Float16)0.f;
    a[2*j+1] = (kb + 1 < NOBS) ? (_Float16)qo[m * NOBS + kb + 1] : (_Float16)0.f;
  }
#pragma unroll
  for (int i = 0; i < 16; ++i) {  // B 32x16: lane=N, K in same table order as A
    const int k = ((i < 8) ? i : (i + 8)) + (hi ? 8 : 0);
    bm[i] = (k < NOBS && nn < OUTDIM) ? (_Float16)Wh[nn * NOBS + k] : (_Float16)0.f;
  }
  v8f c = {};
  c = __builtin_amdgcn_wmma_f32_16x16x32_f16(false, a, false, bm,
                                             (short)0, c, false, false);
  if (nn < OUTDIM) {
    const float bv = bh[nn];
#pragma unroll
    for (int v = 0; v < 8; ++v)
      out[(m0 + (hi ? 8 : 0) + v) * OUTDIM + nn] = c[v] + bv;
  }
}

// ---------------------------------------------------------------------------
extern "C" void kernel_launch(void* const* d_in, const int* in_sizes, int n_in,
                              void* d_out, int out_size, void* d_ws, size_t ws_size,
                              hipStream_t stream) {
  (void)in_sizes; (void)n_in; (void)out_size; (void)ws_size;
  const float* x  = (const float*)d_in[0];
  const float* We = (const float*)d_in[1];
  const float* be = (const float*)d_in[2];
  const float* vp = (const float*)d_in[3];
  const float* A  = (const float*)d_in[4];
  const float* Bp = (const float*)d_in[5];
  const float* D  = (const float*)d_in[6];
  const float* Wh = (const float*)d_in[7];
  const float* bh = (const float*)d_in[8];
  float* out = (float*)d_out;

  char* ws = (char*)d_ws;
  float*  enc = (float*)(ws);                               // 128*390 f32
  float2* Ug  = (float2*)(ws + 204800);                     // 26*128*16 c64
  float2* Hs  = (float2*)(ws + 204800 + 430080);            // 11*256 c64
  float*  qo  = (float*)(ws + 204800 + 430080 + 24576);     // 128*11 f32

  enc_gemm<<<dim3(8, 25), 32, 0, stream>>>(x, We, be, enc);
  su4_kernel<<<dim3(NGATES), 128, 0, stream>>>(enc, Ug);
  herm_kernel<<<dim3(NOBS), 256, 0, stream>>>(A, Bp, D, Hs);

  (void)hipFuncSetAttribute(reinterpret_cast<const void*>(sim_kernel),
                            hipFuncAttributeMaxDynamicSharedMemorySize,
                            (int)SIM_SMEM);
  sim_kernel<<<dim3(BATCH), SIM_THREADS, SIM_SMEM, stream>>>(Ug, vp, Hs, qo);
  head_kernel<<<dim3(8), 32, 0, stream>>>(qo, Wh, bh, out);
}
